// FlowMatchingMoE_36876589204274
// MI455X (gfx1250) — matrix operand order repaired and checked by reading, not compiled, not run on hardware
//
#include <hip/hip_runtime.h>
#include <hip/hip_bf16.h>

// ---- problem constants (match reference) ----
#define B_    4
#define S_    512
#define D_    512
#define H_    1024
#define T_    64
#define E_    8
#define STEPS_ 4
#define N_    (B_ * S_)       // 2048 tokens
#define CIN1  (D_ + T_)       // 576

// ---- GEMM tiling ----
#define BM 128
#define BN 128
#define BK 32
#define LDSS 40               // padded halves per LDS row (80B, 16B aligned rows)

typedef __bf16 bf16;
typedef __attribute__((ext_vector_type(16))) __bf16 bf16x16;
typedef __attribute__((ext_vector_type(8)))  float  floatx8;
typedef int v4i __attribute__((vector_size(16)));

union Frag16 { float4 f[2]; bf16x16 v; };

#if __has_builtin(__builtin_amdgcn_global_load_async_to_lds_b128)
#define HAVE_ASYNC_LDS 1
#else
#define HAVE_ASYNC_LDS 0
#endif

#if HAVE_ASYNC_LDS
__device__ __forceinline__ void async_copy16(const bf16* g, bf16* l)
{
    __builtin_amdgcn_global_load_async_to_lds_b128(
        (__attribute__((address_space(1))) v4i*)g,
        (__attribute__((address_space(3))) v4i*)l,
        0, 0);
}
__device__ __forceinline__ void wait_async_le4()
{
#if __has_builtin(__builtin_amdgcn_s_wait_asynccnt)
    __builtin_amdgcn_s_wait_asynccnt(4);
#else
    asm volatile("s_wait_asynccnt 4" ::: "memory");
#endif
}
__device__ __forceinline__ void wait_async_0()
{
#if __has_builtin(__builtin_amdgcn_s_wait_asynccnt)
    __builtin_amdgcn_s_wait_asynccnt(0);
#else
    asm volatile("s_wait_asynccnt 0" ::: "memory");
#endif
}
#endif

// ============================================================================
// Tiled bf16 WMMA GEMM:  Out[e][n][h] = epilogue( sum_c A[e][n][c]*W[e][h][c] + b[e][h] )
// EPI==0: SiLU -> bf16 Zout ;  EPI==1: Xt[e][n][h] += z * dt (f32)
// Double-buffered LDS with async global->LDS prefetch (CDNA5 ASYNCcnt path).
// ============================================================================
template<int EPI>
__global__ __launch_bounds__(256)
void gemm_bf16_wmma(const bf16* __restrict__ Act,   // (E, Ntok, Cin) bf16
                    const bf16* __restrict__ Wt,    // (E, Nout, Cin) bf16
                    const float* __restrict__ bias, // (E, Nout)
                    bf16* __restrict__ Zout,        // (E, Ntok, Nout) if EPI==0
                    float* __restrict__ Xt,         // (E, Ntok, Nout) if EPI==1
                    int Ntok, int Cin, int Nout, float dtv)
{
    __shared__ bf16 sA[2][BM * LDSS];
    __shared__ bf16 sB[2][BN * LDSS];

    const int e    = blockIdx.z;
    const int bm   = blockIdx.x * BM;
    const int bn   = blockIdx.y * BN;
    const int tid  = threadIdx.x;
    const int lane = tid & 31;
    const int wid  = tid >> 5;       // 0..7
    const int waveM = wid >> 2;      // 0..1  (64 rows each)
    const int waveN = wid & 3;       // 0..3  (32 cols each)
    const int half  = lane >> 4;     // 0/1
    const int l15   = lane & 15;

    const bf16* A = Act + (size_t)e * Ntok * Cin;
    const bf16* W = Wt  + (size_t)e * Nout * Cin;

    // per-thread staging coordinates: 512 16B-chunks per 128x32 tile, 2/thread
    const int row0 = tid >> 2;                 // chunk tid
    const int col0 = (tid & 3) * 8;
    const int row1 = (tid + 256) >> 2;         // chunk tid+256
    const int col1 = ((tid + 256) & 3) * 8;

    floatx8 acc[4][2];
#pragma unroll
    for (int i = 0; i < 4; ++i)
#pragma unroll
        for (int j = 0; j < 2; ++j)
            acc[i][j] = (floatx8){0.f,0.f,0.f,0.f,0.f,0.f,0.f,0.f};

    const int KT = Cin / BK;

#if HAVE_ASYNC_LDS
    // prefetch tile 0 into buffer 0 (4 async ops per wave)
    async_copy16(A + (size_t)(bm + row0) * Cin + col0, &sA[0][row0 * LDSS + col0]);
    async_copy16(A + (size_t)(bm + row1) * Cin + col1, &sA[0][row1 * LDSS + col1]);
    async_copy16(W + (size_t)(bn + row0) * Cin + col0, &sB[0][row0 * LDSS + col0]);
    async_copy16(W + (size_t)(bn + row1) * Cin + col1, &sB[0][row1 * LDSS + col1]);
#endif

    for (int kt = 0; kt < KT; ++kt) {
        const int cur = kt & 1;
        __syncthreads();   // close previous iteration's LDS reads
#if HAVE_ASYNC_LDS
        if (kt + 1 < KT) {
            const int nxt = (kt + 1) & 1;
            const int kc  = (kt + 1) * BK;
            async_copy16(A + (size_t)(bm + row0) * Cin + kc + col0, &sA[nxt][row0 * LDSS + col0]);
            async_copy16(A + (size_t)(bm + row1) * Cin + kc + col1, &sA[nxt][row1 * LDSS + col1]);
            async_copy16(W + (size_t)(bn + row0) * Cin + kc + col0, &sB[nxt][row0 * LDSS + col0]);
            async_copy16(W + (size_t)(bn + row1) * Cin + kc + col1, &sB[nxt][row1 * LDSS + col1]);
            wait_async_le4();   // stage kt complete; stage kt+1 still in flight
        } else {
            wait_async_0();
        }
#else
        {
            const int kc = kt * BK;
            const float4 a0 = *reinterpret_cast<const float4*>(A + (size_t)(bm + row0) * Cin + kc + col0);
            const float4 a1 = *reinterpret_cast<const float4*>(A + (size_t)(bm + row1) * Cin + kc + col1);
            const float4 b0 = *reinterpret_cast<const float4*>(W + (size_t)(bn + row0) * Cin + kc + col0);
            const float4 b1 = *reinterpret_cast<const float4*>(W + (size_t)(bn + row1) * Cin + kc + col1);
            *reinterpret_cast<float4*>(&sA[cur][row0 * LDSS + col0]) = a0;
            *reinterpret_cast<float4*>(&sA[cur][row1 * LDSS + col1]) = a1;
            *reinterpret_cast<float4*>(&sB[cur][row0 * LDSS + col0]) = b0;
            *reinterpret_cast<float4*>(&sB[cur][row1 * LDSS + col1]) = b1;
        }
#endif
        __syncthreads();   // stage kt visible to all waves

        // B fragments: lane holds col n=l15, K = half*16 .. half*16+15 (contiguous)
        Frag16 bfr[2];
#pragma unroll
        for (int nf = 0; nf < 2; ++nf) {
            const float4* pb = reinterpret_cast<const float4*>(
                &sB[cur][(waveN * 32 + nf * 16 + l15) * LDSS + half * 16]);
            bfr[nf].f[0] = pb[0];
            bfr[nf].f[1] = pb[1];
        }
        // A fragments: lane holds row m=l15, K = half*16 .. half*16+15 (contiguous)
#pragma unroll
        for (int mf = 0; mf < 4; ++mf) {
            Frag16 afr;
            const float4* pa = reinterpret_cast<const float4*>(
                &sA[cur][(waveM * 64 + mf * 16 + l15) * LDSS + half * 16]);
            afr.f[0] = pa[0];
            afr.f[1] = pa[1];   // +8 halves
#pragma unroll
            for (int nf = 0; nf < 2; ++nf) {
                acc[mf][nf] = __builtin_amdgcn_wmma_f32_16x16x32_bf16(
                    false, afr.v, false, bfr[nf].v,
                    (short)0, acc[mf][nf], false, false);
            }
        }
    }

    // epilogue: C/D layout: VGPR r -> M = r + 8*half ; N = l15
    const float* bvec = bias + (size_t)e * Nout;
#pragma unroll
    for (int mf = 0; mf < 4; ++mf) {
#pragma unroll
        for (int nf = 0; nf < 2; ++nf) {
            const int feat = bn + waveN * 32 + nf * 16 + l15;
            const float bb = bvec[feat];
#pragma unroll
            for (int r = 0; r < 8; ++r) {
                const int tok = bm + waveM * 64 + mf * 16 + half * 8 + r;
                const float z = acc[mf][nf][r] + bb;
                if (EPI == 0) {
                    const float s = z / (1.0f + __expf(-z));   // SiLU
                    Zout[((size_t)e * Ntok + tok) * Nout + feat] = (bf16)s;
                } else {
                    const size_t idx = ((size_t)e * Ntok + tok) * Nout + feat;
                    Xt[idx] += z * dtv;
                }
            }
        }
    }
}

// ============================================================================
// Row LayerNorm over H_=1024, bf16 in/out, f32 math. One block per row.
// ============================================================================
__global__ __launch_bounds__(256)
void layernorm_bf16(bf16* __restrict__ Z, const float* __restrict__ gamma,
                    const float* __restrict__ beta, int Ntok)
{
    __shared__ float red[256];
    const int row = blockIdx.x;             // 0 .. E*Ntok-1
    const int e   = row / Ntok;
    const int tid = threadIdx.x;
    bf16* zr = Z + (size_t)row * H_;

    float v[4], s = 0.f, ss = 0.f;
#pragma unroll
    for (int i = 0; i < 4; ++i) {
        const float x = (float)zr[tid + i * 256];
        v[i] = x; s += x; ss += x * x;
    }
    red[tid] = s; __syncthreads();
    for (int off = 128; off > 0; off >>= 1) {
        if (tid < off) red[tid] += red[tid + off];
        __syncthreads();
    }
    const float mu = red[0] / (float)H_;
    __syncthreads();
    red[tid] = ss; __syncthreads();
    for (int off = 128; off > 0; off >>= 1) {
        if (tid < off) red[tid] += red[tid + off];
        __syncthreads();
    }
    const float var  = red[0] / (float)H_ - mu * mu;
    const float rstd = rsqrtf(var + 1e-5f);
    const float* g = gamma + (size_t)e * H_;
    const float* b = beta  + (size_t)e * H_;
#pragma unroll
    for (int i = 0; i < 4; ++i) {
        const int c = tid + i * 256;
        zr[c] = (bf16)((v[i] - mu) * rstd * g[c] + b[c]);
    }
}

// ============================================================================
// Router: logits, softmax, top-2 combine weights, per-token aux terms
// ============================================================================
__global__ __launch_bounds__(256)
void router_kernel(const float* __restrict__ x,   // (N, D)
                   const float* __restrict__ gw,  // (E, D)
                   float* __restrict__ comb,      // (N, E)
                   float* __restrict__ zl,        // (N)
                   float* __restrict__ pbuf)      // (N, E)
{
    __shared__ float sgw[E_ * D_];                 // 16 KB
    const int tid = threadIdx.x;
    for (int i = tid; i < E_ * D_; i += 256) sgw[i] = gw[i];
    __syncthreads();

    const int n = blockIdx.x * 256 + tid;
    const float* xr = x + (size_t)n * D_;
    float logit[E_];
#pragma unroll
    for (int e = 0; e < E_; ++e) logit[e] = 0.f;
    for (int c = 0; c < D_; ++c) {
        const float xv = xr[c];
#pragma unroll
        for (int e = 0; e < E_; ++e) logit[e] += xv * sgw[e * D_ + c];
    }
    float m = logit[0];
#pragma unroll
    for (int e = 1; e < E_; ++e) m = fmaxf(m, logit[e]);
    float p[E_], ssum = 0.f;
#pragma unroll
    for (int e = 0; e < E_; ++e) { p[e] = __expf(logit[e] - m); ssum += p[e]; }
    const float lse = m + __logf(ssum);
    zl[n] = lse * lse;
#pragma unroll
    for (int e = 0; e < E_; ++e) { p[e] /= ssum; pbuf[n * E_ + e] = p[e]; }
    // top-2 (first occurrence on ties, like lax.top_k)
    int i1 = 0; float p1 = p[0];
#pragma unroll
    for (int e = 1; e < E_; ++e) if (p[e] > p1) { p1 = p[e]; i1 = e; }
    int i2 = 0; float p2 = -1.f;
#pragma unroll
    for (int e = 0; e < E_; ++e) if (e != i1 && p[e] > p2) { p2 = p[e]; i2 = e; }
    const float denom = p1 + p2 + 1e-9f;
#pragma unroll
    for (int e = 0; e < E_; ++e) comb[n * E_ + e] = 0.f;
    comb[n * E_ + i1] = p1 / denom;
    comb[n * E_ + i2] = p2 / denom;
}

// deterministic single-block reduction of aux losses
__global__ __launch_bounds__(256)
void aux_finalize(const float* __restrict__ zl, const float* __restrict__ pbuf,
                  float* __restrict__ aux_out)
{
    __shared__ float red[256];
    const int tid = threadIdx.x;
    float acc[9];
#pragma unroll
    for (int v = 0; v < 9; ++v) acc[v] = 0.f;
    for (int n = tid; n < N_; n += 256) {
        acc[0] += zl[n];
#pragma unroll
        for (int e = 0; e < E_; ++e) acc[1 + e] += pbuf[n * E_ + e];
    }
    float results[9];
    for (int v = 0; v < 9; ++v) {
        red[tid] = acc[v]; __syncthreads();
        for (int off = 128; off > 0; off >>= 1) {
            if (tid < off) red[tid] += red[tid + off];
            __syncthreads();
        }
        results[v] = red[0];
        __syncthreads();
    }
    if (tid == 0) {
        const float z_loss = results[0] / (float)N_;
        float lb = 0.f;
        for (int e = 0; e < E_; ++e) {
            const float mp = results[1 + e] / (float)N_;
            const float dd = mp - 1.0f / (float)E_;
            lb += dd * dd;
        }
        lb *= (float)E_;
        aux_out[0] = 0.001f * z_loss + 0.01f * lb;
    }
}

// ---- small utility kernels ----
__global__ void cast_f32_to_bf16(const float* __restrict__ in, bf16* __restrict__ out, int n)
{
    const int i = blockIdx.x * 256 + threadIdx.x;
    if (i < n) out[i] = (bf16)in[i];
}

__global__ void init_xt_kernel(const float* __restrict__ x, float* __restrict__ Xt)
{
    const int i = blockIdx.x * 256 + threadIdx.x;   // over E*N*D
    const int nd = i % (N_ * D_);
    Xt[i] = x[nd];
}

__global__ void build_hin_kernel(const float* __restrict__ Xt, bf16* __restrict__ Hin, float t)
{
    const int i  = blockIdx.x * 256 + threadIdx.x;  // over E*N*CIN1
    const int c  = i % CIN1;
    const int en = i / CIN1;                        // e*N + n
    float v;
    if (c < D_) {
        v = Xt[(size_t)en * D_ + c];
    } else {
        const int j  = c - D_;                      // 0..63
        const int jj = j & 31;                      // freq index
        const float freq = __expf((float)jj * (-logf(10000.f) / 31.f));
        const float a = t * freq;
        v = (j < 32) ? __sinf(a) : __cosf(a);
    }
    Hin[i] = (bf16)v;
}

__global__ void combine_kernel(const float* __restrict__ comb, const float* __restrict__ Xt,
                               float* __restrict__ out)
{
    const int i = blockIdx.x * 256 + threadIdx.x;   // over N*D
    const int n = i / D_;
    const int d = i % D_;
    float s = 0.f;
#pragma unroll
    for (int e = 0; e < E_; ++e)
        s += comb[n * E_ + e] * Xt[((size_t)e * N_ + n) * D_ + d];
    out[i] = s;
}

// ============================================================================
extern "C" void kernel_launch(void* const* d_in, const int* in_sizes, int n_in,
                              void* d_out, int out_size, void* d_ws, size_t ws_size,
                              hipStream_t stream)
{
    (void)in_sizes; (void)n_in; (void)out_size; (void)ws_size;

    const float* x   = (const float*)d_in[0];
    const float* gw  = (const float*)d_in[1];
    const float* W1  = (const float*)d_in[2];
    const float* b1  = (const float*)d_in[3];
    const float* g1v = (const float*)d_in[4];
    const float* be1 = (const float*)d_in[5];
    const float* W2  = (const float*)d_in[6];
    const float* b2  = (const float*)d_in[7];
    const float* g2v = (const float*)d_in[8];
    const float* be2 = (const float*)d_in[9];
    const float* W3  = (const float*)d_in[10];
    const float* b3  = (const float*)d_in[11];
    float* out = (float*)d_out;

    // ---- workspace partition ----
    char* ws = (char*)d_ws;
    size_t off = 0;
    auto alloc = [&](size_t bytes) -> void* {
        void* p = ws + off;
        off = (off + bytes + 255) & ~(size_t)255;
        return p;
    };
    float* comb = (float*)alloc((size_t)N_ * E_ * 4);
    float* zl   = (float*)alloc((size_t)N_ * 4);
    float* pbuf = (float*)alloc((size_t)N_ * E_ * 4);
    bf16*  Wb1  = (bf16*)alloc((size_t)E_ * H_ * CIN1 * 2);
    bf16*  Wb2  = (bf16*)alloc((size_t)E_ * H_ * H_ * 2);
    bf16*  Wb3  = (bf16*)alloc((size_t)E_ * D_ * H_ * 2);
    float* Xt   = (float*)alloc((size_t)E_ * N_ * D_ * 4);
    bf16*  Hin  = (bf16*)alloc((size_t)E_ * N_ * CIN1 * 2);
    bf16*  Z1   = (bf16*)alloc((size_t)E_ * N_ * H_ * 2);
    bf16*  Z2   = (bf16*)alloc((size_t)E_ * N_ * H_ * 2);

    // ---- router + aux ----
    router_kernel<<<N_ / 256, 256, 0, stream>>>(x, gw, comb, zl, pbuf);
    aux_finalize<<<1, 256, 0, stream>>>(zl, pbuf, out + (size_t)N_ * D_);

    // ---- weight conversion (once per launch) ----
    const int n1 = E_ * H_ * CIN1;
    const int n2 = E_ * H_ * H_;
    const int n3 = E_ * D_ * H_;
    cast_f32_to_bf16<<<(n1 + 255) / 256, 256, 0, stream>>>(W1, Wb1, n1);
    cast_f32_to_bf16<<<(n2 + 255) / 256, 256, 0, stream>>>(W2, Wb2, n2);
    cast_f32_to_bf16<<<(n3 + 255) / 256, 256, 0, stream>>>(W3, Wb3, n3);

    // ---- x0 broadcast ----
    init_xt_kernel<<<(E_ * N_ * D_) / 256, 256, 0, stream>>>(x, Xt);

    // ---- Euler integration over STEPS ----
    const float dt = 1.0f / (float)STEPS_;
    const dim3 grid12(N_ / BM, H_ / BN, E_);
    const dim3 grid3 (N_ / BM, D_ / BN, E_);
    for (int sstep = 0; sstep < STEPS_; ++sstep) {
        const float t = (float)sstep * dt;
        build_hin_kernel<<<(E_ * N_ * CIN1) / 256, 256, 0, stream>>>(Xt, Hin, t);
        gemm_bf16_wmma<0><<<grid12, 256, 0, stream>>>(Hin, Wb1, b1, Z1, nullptr,
                                                      N_, CIN1, H_, 0.f);
        layernorm_bf16<<<E_ * N_, 256, 0, stream>>>(Z1, g1v, be1, N_);
        gemm_bf16_wmma<0><<<grid12, 256, 0, stream>>>(Z1, Wb2, b2, Z2, nullptr,
                                                      N_, H_, H_, 0.f);
        layernorm_bf16<<<E_ * N_, 256, 0, stream>>>(Z2, g2v, be2, N_);
        gemm_bf16_wmma<1><<<grid3, 256, 0, stream>>>(Z2, Wb3, b3, nullptr, Xt,
                                                     N_, H_, D_, dt);
    }

    // ---- weighted combine over experts ----
    combine_kernel<<<(N_ * D_) / 256, 256, 0, stream>>>(comb, Xt, out);
}
